// Net_24773371364030
// MI455X (gfx1250) — matrix operand order repaired and checked by reading, not compiled, not run on hardware
//
#include <hip/hip_runtime.h>
#include <hip/hip_bf16.h>

#define BB  4096
#define NUU 256

typedef __attribute__((ext_vector_type(16))) _Float16 v16h;
typedef __attribute__((ext_vector_type(8)))  _Float16 v8h;
typedef __attribute__((ext_vector_type(8)))  float    v8f;

__device__ __forceinline__ float lrelu(float v) { return v > 0.f ? v : 0.01f * v; }

// ---------------------------------------------------------------------------
// WMMA fragment helpers (16x16x32 f16; contiguous k = 16*(lane/16)+s in both A and B)
// ---------------------------------------------------------------------------
// A: 16x32 f16, row-major in LDS. lane<16: M=lane, elems0-15=K0..15
//                                 lane>=16: M=lane-16, K16..31.
__device__ __forceinline__ v16h load_fragA(const _Float16* base, int lane) {
    const int r    = lane & 15;
    const int koff = (lane & 16) ? 16 : 0;
    const _Float16* p = base + r * 32 + koff;
    v8h lo = *(const v8h*)(p);
    v8h hi = *(const v8h*)(p + 8);
    v16h out;
#pragma unroll
    for (int i = 0; i < 8; ++i) { out[i] = lo[i]; out[i + 8] = hi[i]; }
    return out;
}

// B: 32x16 f16. Weight tile stored N-major [16][32] (i.e. W^T rows = outputs).
// lane: N=lane&15; K-run = 0..15 (lanes 0-15) or 16..31 (lanes 16-31), contiguous.
__device__ __forceinline__ v16h load_fragB(const _Float16* baseNmajor, int lane) {
    const int n  = lane & 15;
    const int kb = (lane & 16) ? 16 : 0;
    return *(const v16h*)(baseNmajor + n * 32 + kb);
}

__device__ __forceinline__ v8f wmma_f16(v16h a, v16h b, v8f c) {
    return __builtin_amdgcn_wmma_f32_16x16x32_f16(false, a, false, b, (short)0, c,
                                                  false, false);
}

// ---------------------------------------------------------------------------
// Kernel A: per-batch-row path (MLP + convs + GRU + g2u), 1 wave per row.
// ---------------------------------------------------------------------------
struct ParamsA {
    const float *gs, *mm, *hidden;
    const float *W0, *b0, *W1, *b1, *W2, *b2, *W3lin, *b3lin;
    const float *Wih, *Whh, *bih, *bhh;
    const float *Wg2u, *bg2u, *Wc2g, *bc2g;
    const float *bu2, *bc2u;
    const float *c1W, *c1b, *c2W, *c2b, *c22W, *c22b, *c23W, *c23b, *c3W, *c3b;
    float *h_out, *ws_m, *ws_g2u;
};

__device__ void conv3x3(const float* __restrict__ W, const float* __restrict__ bias,
                        const float* src, float* dst, int IC, int OC, int lane) {
    for (int p = lane; p < 100; p += 32) {
        const int x = p % 10, y = p / 10;
        for (int oc = 0; oc < OC; ++oc) {
            float acc = bias[oc];
            for (int ic = 0; ic < IC; ++ic) {
                const float* wp = W + (oc * IC + ic) * 9;
                const float* sp = src + ic * 100;
                for (int dy = -1; dy <= 1; ++dy) {
                    const int yy = y + dy;
                    if ((unsigned)yy >= 10u) continue;
                    for (int dx = -1; dx <= 1; ++dx) {
                        const int xx = x + dx;
                        if ((unsigned)xx >= 10u) continue;
                        acc += wp[(dy + 1) * 3 + (dx + 1)] * sp[yy * 10 + xx];
                    }
                }
            }
            dst[oc * 100 + p] = lrelu(acc);
        }
    }
}

__global__ __launch_bounds__(128) void fused_global_kernel(ParamsA P) {
    __shared__ float Pb[4][1200];
    __shared__ float Qb[4][1200];
    __shared__ float sx0[4][32], sx1[4][64], sx2[4][64], sx3[4][64];
    __shared__ float sgi[4][384], sgh[4][384], sh[4][128];

    const int tid  = threadIdx.x;
    const int w    = tid >> 5;
    const int lane = tid & 31;
    const int row  = blockIdx.x * 4 + w;
    float* Pr = Pb[w];
    float* Qr = Qb[w];

    // minimap (10x10x10) -> LDS
    const float* mmp = P.mm + (size_t)row * 1000;
    for (int i = lane; i < 1000; i += 32) Pr[i] = mmp[i];
    __syncthreads();

    // conv1 1x1, 10 -> 12
    for (int p = lane; p < 100; p += 32) {
        for (int oc = 0; oc < 12; ++oc) {
            float acc = P.c1b[oc];
            for (int ic = 0; ic < 10; ++ic) acc += P.c1W[oc * 10 + ic] * Pr[ic * 100 + p];
            Qr[oc * 100 + p] = lrelu(acc);
        }
    }
    __syncthreads();
    conv3x3(P.c2W,  P.c2b,  Qr, Pr, 12, 12, lane); __syncthreads();
    conv3x3(P.c22W, P.c22b, Pr, Qr, 12, 6,  lane); __syncthreads();
    conv3x3(P.c23W, P.c23b, Qr, Pr, 6,  6,  lane); __syncthreads();
    conv3x3(P.c3W,  P.c3b,  Pr, Qr, 6,  6,  lane); __syncthreads();

    // export m (6x100) for the per-unit gather
    for (int i = lane; i < 600; i += 32) P.ws_m[(size_t)row * 600 + i] = Qr[i];

    // lin0: 97 -> 32
    const float* gsr = P.gs + (size_t)row * 97;
    {
        const int o = lane;
        float acc = P.b0[o];
        for (int k = 0; k < 97; ++k) acc += P.W0[o * 97 + k] * gsr[k];
        sx0[w][o] = lrelu(acc);
    }
    __syncthreads();
    // lin1: 32 -> 64
    for (int j = 0; j < 2; ++j) {
        const int o = lane + 32 * j;
        float acc = P.b1[o];
        for (int k = 0; k < 32; ++k) acc += P.W1[o * 32 + k] * sx0[w][k];
        sx1[w][o] = lrelu(acc);
    }
    __syncthreads();
    // lin2(x) + lin_c2g(m3)
    for (int j = 0; j < 2; ++j) {
        const int o = lane + 32 * j;
        float acc = P.b2[o] + P.bc2g[o];
        for (int k = 0; k < 64; ++k) acc += P.W2[o * 64 + k] * sx1[w][k];
        const float* Wc = P.Wc2g + (size_t)o * 600;
        for (int k = 0; k < 600; ++k) acc += Wc[k] * Qr[k];
        sx2[w][o] = lrelu(acc);
    }
    __syncthreads();
    // lin3: 64 -> 64
    for (int j = 0; j < 2; ++j) {
        const int o = lane + 32 * j;
        float acc = P.b3lin[o];
        for (int k = 0; k < 64; ++k) acc += P.W3lin[o * 64 + k] * sx2[w][k];
        sx3[w][o] = lrelu(acc);
    }
    __syncthreads();
    // GRU gates
    const float* hin = P.hidden + (size_t)row * 128;
    for (int j = 0; j < 12; ++j) {
        const int o = lane + 32 * j;
        float gi = P.bih[o], gh = P.bhh[o];
        for (int k = 0; k < 64; ++k)  gi += P.Wih[o * 64 + k]  * sx3[w][k];
        for (int k = 0; k < 128; ++k) gh += P.Whh[o * 128 + k] * hin[k];
        sgi[w][o] = gi; sgh[w][o] = gh;
    }
    __syncthreads();
    for (int j = 0; j < 4; ++j) {
        const int i = lane + 32 * j;
        const float rg = 1.f / (1.f + expf(-(sgi[w][i]       + sgh[w][i])));
        const float zg = 1.f / (1.f + expf(-(sgi[w][128 + i] + sgh[w][128 + i])));
        const float ng = tanhf(sgi[w][256 + i] + rg * sgh[w][256 + i]);
        const float hv = (1.f - zg) * ng + zg * hin[i];
        sh[w][i] = hv;
        P.h_out[(size_t)row * 128 + i] = hv;
    }
    __syncthreads();
    // g2u = W_g2u*h + (b_g2u + b_u2 + b_c2u)  (all per-unit biases folded here)
    {
        const int o = lane;
        float acc = P.bg2u[o] + P.bu2[o] + P.bc2u[o];
        for (int k = 0; k < 128; ++k) acc += P.Wg2u[o * 128 + k] * sh[w][k];
        P.ws_g2u[(size_t)row * 32 + o] = acc;
    }
}

// ---------------------------------------------------------------------------
// Kernel B: per-unit MLP on WMMA. 1 wave per 16-unit tile, 8 waves/block.
// ---------------------------------------------------------------------------
struct ParamsB {
    const float *rawUnits, *emb;
    const int   *coords;
    const float *Wu1, *bu1, *Wu2, *Wc2u, *Wu22, *bu22, *Wu23, *bu23, *Wu3, *bu3;
    const float *ws_m, *ws_g2u;
    float *out;
};

__global__ __launch_bounds__(256) void unit_mlp_wmma(ParamsB P) {
    __shared__ __align__(16) _Float16 sWu1[16 * 32];
    __shared__ __align__(16) _Float16 sWu2[32 * 32];
    __shared__ __align__(16) _Float16 sWu22[16 * 32];
    __shared__ __align__(16) _Float16 sWu23[16 * 32];
    __shared__ float sB1[16], sB22[16], sB23[16], sW3[32], sB3[2];
    __shared__ __align__(16) _Float16 sA[8][2][16 * 32];   // ping/pong A-tiles per wave
    __shared__ float sS[8][16 * 16];                       // final f32 stage per wave

    const int tid  = threadIdx.x;
    const int w    = tid >> 5;
    const int lane = tid & 31;

    // Stage weights to LDS as f16, N-major [out][32] with K zero-padding.
    for (int i = tid; i < 16 * 32; i += 256) {
        const int o = i >> 5, k = i & 31;
        sWu1[i]  = (_Float16)(k < 29 ? P.Wu1[o * 29 + k] : 0.f);
        sWu22[i] = (_Float16)(P.Wu22[o * 32 + k]);
        sWu23[i] = (_Float16)(k < 16 ? P.Wu23[o * 16 + k] : 0.f);
    }
    // Combined [lin_u2 | lin_c2u] weights: K = [0..15]=u1-out, [16..21]=mm-info.
    for (int i = tid; i < 32 * 32; i += 256) {
        const int o = i >> 5, k = i & 31;
        const float v = (k < 16) ? P.Wu2[o * 16 + k]
                                 : ((k < 22) ? P.Wc2u[o * 6 + (k - 16)] : 0.f);
        sWu2[i] = (_Float16)v;
    }
    if (tid < 16) { sB1[tid] = P.bu1[tid]; sB22[tid] = P.bu22[tid]; sB23[tid] = P.bu23[tid]; }
    if (tid < 32) sW3[tid] = P.Wu3[tid];
    if (tid < 2)  sB3[tid] = P.bu3[tid];
    __syncthreads();

    const int tile = blockIdx.x * 8 + w;
    const int b    = tile >> 4;
    const int u0   = (tile & 15) << 4;
    _Float16* A0 = &sA[w][0][0];
    _Float16* A1 = &sA[w][1][0];
    float*    S  = &sS[w][0];

    // Stage 0: build A0 = [emb(8) | raw(21) | pad(3)]; A1 cols16..31 = [mmInfo(6)|0]
    if (lane < 16) {
        const int u = u0 + lane;
        const float* ru = P.rawUnits + ((size_t)b * NUU + u) * 22;
        const int ty = (int)ru[0];
        const float* er = P.emb + ty * 8;
        for (int k = 0; k < 8; ++k)  A0[lane * 32 + k]     = (_Float16)er[k];
        for (int k = 0; k < 21; ++k) A0[lane * 32 + 8 + k] = (_Float16)ru[1 + k];
        A0[lane * 32 + 29] = (_Float16)0.f;
        A0[lane * 32 + 30] = (_Float16)0.f;
        A0[lane * 32 + 31] = (_Float16)0.f;
        const int* cp = P.coords + ((size_t)b * NUU + u) * 2;
        const int idx = cp[0] * 10 + cp[1];
        const float* mrow = P.ws_m + (size_t)b * 600 + idx;
        for (int j = 0; j < 6; ++j) A1[lane * 32 + 16 + j] = (_Float16)mrow[j * 100];
        for (int k = 22; k < 32; ++k) A1[lane * 32 + k] = (_Float16)0.f;
    }
    __syncthreads();

    const int n  = lane & 15;
    const int mb = (lane & 16) ? 8 : 0;

    // Stage 1: lin_u1 (29 -> 16), write activations to A1 cols 0..15
    {
        v16h a  = load_fragA(A0, lane);
        v16h bw = load_fragB(sWu1, lane);
        v8f c = {0.f, 0.f, 0.f, 0.f, 0.f, 0.f, 0.f, 0.f};
        c = wmma_f16(a, bw, c);
        const float bias = sB1[n];
#pragma unroll
        for (int i = 0; i < 8; ++i)
            A1[(mb + i) * 32 + n] = (_Float16)lrelu(c[i] + bias);
    }
    __syncthreads();

    // Stage 2: [lin_u2 | lin_c2u] (22 -> 32) + g2u broadcast, write A0 (32 cols)
    {
        v16h a  = load_fragA(A1, lane);
        v16h b0 = load_fragB(sWu2, lane);
        v16h b1 = load_fragB(sWu2 + 16 * 32, lane);
        v8f c0 = {0.f, 0.f, 0.f, 0.f, 0.f, 0.f, 0.f, 0.f};
        v8f c1 = {0.f, 0.f, 0.f, 0.f, 0.f, 0.f, 0.f, 0.f};
        c0 = wmma_f16(a, b0, c0);
        c1 = wmma_f16(a, b1, c1);
        const float g0 = P.ws_g2u[(size_t)b * 32 + n];
        const float g1 = P.ws_g2u[(size_t)b * 32 + 16 + n];
#pragma unroll
        for (int i = 0; i < 8; ++i) {
            A0[(mb + i) * 32 + n]      = (_Float16)lrelu(c0[i] + g0);
            A0[(mb + i) * 32 + 16 + n] = (_Float16)lrelu(c1[i] + g1);
        }
    }
    __syncthreads();

    // Stage 3: lin_u2_2 (32 -> 16), write A1 cols 0..15, zero cols 16..31
    {
        v16h a  = load_fragA(A0, lane);
        v16h bw = load_fragB(sWu22, lane);
        v8f c = {0.f, 0.f, 0.f, 0.f, 0.f, 0.f, 0.f, 0.f};
        c = wmma_f16(a, bw, c);
        const float bias = sB22[n];
#pragma unroll
        for (int i = 0; i < 8; ++i) {
            A1[(mb + i) * 32 + n]      = (_Float16)lrelu(c[i] + bias);
            A1[(mb + i) * 32 + 16 + n] = (_Float16)0.f;
        }
    }
    __syncthreads();

    // Stage 4: lin_u2_3 (16 -> 16), keep f32 for the final head
    {
        v16h a  = load_fragA(A1, lane);
        v16h bw = load_fragB(sWu23, lane);
        v8f c = {0.f, 0.f, 0.f, 0.f, 0.f, 0.f, 0.f, 0.f};
        c = wmma_f16(a, bw, c);
        const float bias = sB23[n];
#pragma unroll
        for (int i = 0; i < 8; ++i)
            S[(mb + i) * 16 + n] = lrelu(c[i] + bias);
    }
    __syncthreads();

    // Stage 5: lin_u3 (16 -> 2) + log_softmax, one row per lane
    if (lane < 16) {
        const int u = u0 + lane;
        float z0 = sB3[0], z1 = sB3[1];
        for (int k = 0; k < 16; ++k) {
            const float s = S[lane * 16 + k];
            z0 += s * sW3[k];
            z1 += s * sW3[16 + k];
        }
        const float mx = fmaxf(z0, z1), mn = fminf(z0, z1);
        const float l  = log1pf(expf(mn - mx));
        float* op = P.out + ((size_t)b * NUU + u) * 2;
        op[0] = z0 - mx - l;
        op[1] = z1 - mx - l;
    }
}

// ---------------------------------------------------------------------------
extern "C" void kernel_launch(void* const* d_in, const int* in_sizes, int n_in,
                              void* d_out, int out_size, void* d_ws, size_t ws_size,
                              hipStream_t stream) {
    (void)in_sizes; (void)n_in; (void)out_size; (void)ws_size;
    auto f = [&](int i) { return (const float*)d_in[i]; };

    float* ws_m   = (float*)d_ws;                   // B x 600 conv features
    float* ws_g2u = ws_m + (size_t)BB * 600;        // B x 32 fused broadcast+biases

    ParamsA pa;
    pa.gs = f(0); pa.mm = f(1); pa.hidden = f(3);
    pa.W0 = f(6);  pa.b0 = f(7);  pa.W1 = f(8);  pa.b1 = f(9);
    pa.W2 = f(10); pa.b2 = f(11); pa.W3lin = f(12); pa.b3lin = f(13);
    pa.Wih = f(14); pa.Whh = f(15); pa.bih = f(16); pa.bhh = f(17);
    pa.Wg2u = f(28); pa.bg2u = f(29); pa.Wc2g = f(30); pa.bc2g = f(31);
    pa.bu2 = f(21); pa.bc2u = f(33);
    pa.c1W = f(34); pa.c1b = f(35); pa.c2W = f(36); pa.c2b = f(37);
    pa.c22W = f(38); pa.c22b = f(39); pa.c23W = f(40); pa.c23b = f(41);
    pa.c3W = f(42); pa.c3b = f(43);
    pa.h_out = (float*)d_out + (size_t)BB * NUU * 2;
    pa.ws_m = ws_m; pa.ws_g2u = ws_g2u;
    fused_global_kernel<<<BB / 4, 128, 0, stream>>>(pa);

    ParamsB pb;
    pb.rawUnits = f(2); pb.emb = f(5); pb.coords = (const int*)d_in[4];
    pb.Wu1 = f(18); pb.bu1 = f(19); pb.Wu2 = f(20); pb.Wc2u = f(32);
    pb.Wu22 = f(22); pb.bu22 = f(23); pb.Wu23 = f(24); pb.bu23 = f(25);
    pb.Wu3 = f(26); pb.bu3 = f(27);
    pb.ws_m = ws_m; pb.ws_g2u = ws_g2u; pb.out = (float*)d_out;
    unit_mlp_wmma<<<(BB * NUU / 16) / 8, 256, 0, stream>>>(pb);
}
